// FEGCL_46127948759594
// MI455X (gfx1250) — matrix operand order repaired and checked by reading, not compiled, not run on hardware
//
#include <hip/hip_runtime.h>
#include <hip/hip_bf16.h>
#include <math.h>

// ---------------------------------------------------------------------------
// FEGCL fused layer for MI455X (gfx1250, wave32, WMMA bf16->f32)
// Edge pass: 2 waves/block, 32 edges (2 M-tiles) per wave so every B fragment
// fetched from L2 is reused by two WMMA tiles.
// ---------------------------------------------------------------------------

typedef __attribute__((ext_vector_type(16))) __bf16 v16bf;
typedef __attribute__((ext_vector_type(8)))  float  v8f;

#define EWAVES 2            // waves per block, edge kernel
#define NWAVES 4            // waves per block, node kernel
// fragment table offsets (units of one 512-element bf16 fragment)
#define OFF_WE1 0     // 5 kt x 8 nt
#define OFF_WE2 40    // 4 x 8
#define OFF_WC1 72    // 4 x 8
#define OFF_WQ1 104   // 4 x 8
#define OFF_WN1 136   // 6 x 8
#define OFF_WN2 184   // 4 x 4
#define NFRAGS  200

union ABFrag { v16bf v; unsigned int u[8]; uint4 q[2]; };

__device__ __forceinline__ unsigned short f2bf(float x) {
  unsigned int u = __float_as_uint(x);
  u += 0x7FFFu + ((u >> 16) & 1u);            // round-to-nearest-even
  return (unsigned short)(u >> 16);
}
__device__ __forceinline__ unsigned int pack2bf(float a, float b) {
  return (unsigned int)f2bf(a) | ((unsigned int)f2bf(b) << 16);
}
__device__ __forceinline__ float rcpf(float x) { return __builtin_amdgcn_rcpf(x); }
__device__ __forceinline__ float siluf(float x) {
  return x * rcpf(1.0f + __expf(-x));         // v_rcp_f32, not IEEE div chain
}

// A fragment (16x32 bf16 tile) from LDS row-major [rows][ldw], CDNA5 A-layout:
// lane: M=lane&15, half=lane>>4 ; VGPR v pair -> K = 32*kt + 16*(v>>2) + 8*half + 2*(v&3)
__device__ __forceinline__ v16bf load_a_lds(const unsigned short* base, int ldw,
                                            int kt, int lane) {
  ABFrag a;
  const int M = lane & 15, half = lane >> 4;
  const unsigned short* row = base + M * ldw + kt * 32 + 8 * half;
  #pragma unroll
  for (int v = 0; v < 8; ++v) {
    const int k = 16 * (v >> 2) + 2 * (v & 3);
    a.u[v] = *(const unsigned int*)(row + k);   // coalesces to 2x ds_load_b128
  }
  return a.v;
}

// B fragment: pre-swizzled in global scratch, 32B contiguous per lane (L2-resident)
__device__ __forceinline__ v16bf load_b_frag(const unsigned short* frags, int f, int lane) {
  ABFrag b;
  const uint4* p = (const uint4*)(frags + ((size_t)f * 512 + (size_t)lane * 16));
  b.q[0] = p[0];
  b.q[1] = p[1];
  return b.v;
}

__device__ __forceinline__ v8f wmma_bf16(v16bf a, v16bf b, v8f c) {
  return __builtin_amdgcn_wmma_f32_16x16x32_bf16(false, a, false, b, (short)0, c,
                                                 false, false);
}

// ---------------------------------------------------------------------------
__global__ void fegcl_zero(float* __restrict__ p, size_t n) {
  for (size_t i = (size_t)blockIdx.x * blockDim.x + threadIdx.x; i < n;
       i += (size_t)gridDim.x * blockDim.x)
    p[i] = 0.0f;
}

// Convert weights -> bf16 fragments in B-layout: K = 32*kt + 16*(e>>3) + 8*half + (e&7), N = 16*nt + nl
__global__ void fegcl_prep(const float* __restrict__ We1, const float* __restrict__ We2,
                           const float* __restrict__ Wc1, const float* __restrict__ Wq1,
                           const float* __restrict__ Wn1, const float* __restrict__ Wn2,
                           unsigned short* __restrict__ frags) {
  const int f = blockIdx.x;
  const int t = threadIdx.x;              // 512 threads: one per fragment element
  const int lane = t >> 4, e = t & 15;
  const float* W; int base, NT, sK, sN;
  if      (f < OFF_WE2) { W = We1; base = OFF_WE1; NT = 8; sK = 136; sN = 128; }
  else if (f < OFF_WC1) { W = We2; base = OFF_WE2; NT = 8; sK = 128; sN = 128; }
  else if (f < OFF_WQ1) { W = Wc1; base = OFF_WC1; NT = 8; sK = 128; sN = 128; }
  else if (f < OFF_WN1) { W = Wq1; base = OFF_WQ1; NT = 8; sK = 128; sN = 128; }
  else if (f < OFF_WN2) { W = Wn1; base = OFF_WN1; NT = 8; sK = 192; sN = 128; }
  else                  { W = Wn2; base = OFF_WN2; NT = 4; sK = 128; sN = 64;  }
  const int loc = f - base;
  const int kt = loc / NT, nt = loc % NT;
  const int half = lane >> 4, nl = lane & 15;
  const int K = kt * 32 + 16 * (e >> 3) + 8 * half + (e & 7);
  const int N = nt * 16 + nl;
  float v = (K < sK && N < sN) ? W[(size_t)K * sN + N] : 0.0f;
  frags[(size_t)f * 512 + (size_t)lane * 16 + e] = f2bf(v);
}

// ---------------------------------------------------------------------------
// Edge kernel: 2 waves/block (64 threads), 32 edges per wave (2 M-tiles).
__global__ __launch_bounds__(64) void fegcl_edge(
    const float* __restrict__ h, const int* __restrict__ ei,
    const float* __restrict__ coord, const float* __restrict__ quat,
    const float* __restrict__ be1, const float* __restrict__ be2,
    const float* __restrict__ bc1, const float* __restrict__ wc2,
    const float* __restrict__ bq1, const float* __restrict__ Wq2,
    const unsigned short* __restrict__ frags,
    float* __restrict__ agg, float* __restrict__ transAcc,
    float* __restrict__ tanAcc, int E_) {
  __shared__ unsigned int sA[EWAVES][2560];   // 32x160 bf16 (ein) -> 32x128 (edge_feat)
  __shared__ unsigned int sB[EWAVES][2048];   // 32x128 bf16 (hidden1)
  __shared__ int   eRow[EWAVES][32], eCol[EWAVES][32];
  __shared__ float cdS[EWAVES][32][3];
  __shared__ float shc2[128];
  __shared__ float shq2[384];

  const int tid = threadIdx.x, wave = tid >> 5, lane = tid & 31;
  const int nl = lane & 15, half = lane >> 4;
  for (int i = tid; i < 128; i += 64) shc2[i] = wc2[i];
  for (int i = tid; i < 384; i += 64) shq2[i] = Wq2[i];

  const int ebase = (blockIdx.x * EWAVES + wave) * 32;
  unsigned short* einS = (unsigned short*)sA[wave];
  unsigned short* hidS = (unsigned short*)sB[wave];

  // ---- phase 1: per-edge geometric features (all 32 lanes, one edge each) ----
  {
    int e = min(ebase + lane, E_ - 1);
    int r = ei[e], c = ei[E_ + e];
    eRow[wave][lane] = r; eCol[wave][lane] = c;
    float cdx = coord[r*3+0] - coord[c*3+0];
    float cdy = coord[r*3+1] - coord[c*3+1];
    float cdz = coord[r*3+2] - coord[c*3+2];
    cdS[wave][lane][0] = cdx; cdS[wave][lane][1] = cdy; cdS[wave][lane][2] = cdz;
    float radial = cdx*cdx + cdy*cdy + cdz*cdz;
    float invn = rcpf(sqrtf(radial) + 1e-8f);
    float qrx = quat[r*4], qry = quat[r*4+1], qrz = quat[r*4+2], qrw = quat[r*4+3];
    float qcx = quat[c*4], qcy = quat[c*4+1], qcz = quat[c*4+2], qcw = quat[c*4+3];
    float inv = rcpf(qrx*qrx + qry*qry + qrz*qrz + qrw*qrw);
    float px = -qrx*inv, py = -qry*inv, pz = -qrz*inv, pw = qrw*inv;   // qinv
    float qpx = pw*qcx + px*qcw + py*qcz - pz*qcy;                     // qmul(qi,qc)
    float qpy = pw*qcy - px*qcz + py*qcw + pz*qcx;
    float qpz = pw*qcz + px*qcy - py*qcx + pz*qcw;
    float qpw = pw*qcw - px*qcx - py*qcy - pz*qcz;
    float vx = cdx*invn, vy = cdy*invn, vz = cdz*invn;                 // rot_apply
    float tx = 2.f*(py*vz - pz*vy);
    float ty = 2.f*(pz*vx - px*vz);
    float tz = 2.f*(px*vy - py*vx);
    float ux = -(vx + pw*tx + (py*tz - pz*ty));
    float uy = -(vy + pw*ty + (pz*tx - px*tz));
    float uz = -(vz + pw*tz + (px*ty - py*tx));
    unsigned int* ep = (unsigned int*)(einS + lane*160 + 128);
    ep[0] = pack2bf(radial, qpx);
    ep[1] = pack2bf(qpy, qpz);
    ep[2] = pack2bf(qpw, ux);
    ep[3] = pack2bf(uy, uz);
    #pragma unroll
    for (int k = 4; k < 16; ++k) ep[k] = 0u;   // zero-pad K 136..159
  }
  __syncthreads();

  // ---- phase 2: gather h[row], h[col] -> ein (bf16) ----
  for (int i = lane; i < 512; i += 32) {
    int m = i >> 4, f4 = (i & 15) * 4;
    float4 hr = *(const float4*)(h + (size_t)eRow[wave][m]*64 + f4);
    float4 hc = *(const float4*)(h + (size_t)eCol[wave][m]*64 + f4);
    unsigned int* ep = (unsigned int*)(einS + m*160);
    int j = f4 >> 1;
    ep[j+0]    = pack2bf(hr.x, hr.y);
    ep[j+1]    = pack2bf(hr.z, hr.w);
    ep[32+j+0] = pack2bf(hc.x, hc.y);
    ep[32+j+1] = pack2bf(hc.z, hc.w);
  }
  __syncthreads();

  // ---- phase 3: hidden1 = silu(ein @ We1 + be1) ----
  for (int nt = 0; nt < 8; ++nt) {
    v8f acc0 = {}, acc1 = {};
    #pragma unroll
    for (int kt = 0; kt < 5; ++kt) {
      v16bf b = load_b_frag(frags, OFF_WE1 + kt*8 + nt, lane);
      acc0 = wmma_bf16(load_a_lds(einS,           160, kt, lane), b, acc0);
      acc1 = wmma_bf16(load_a_lds(einS + 16*160,  160, kt, lane), b, acc1);
    }
    const float bias = be1[nt*16 + nl];
    #pragma unroll
    for (int r = 0; r < 8; ++r) {
      const int M = r + 8*half;
      hidS[M*128        + nt*16 + nl] = f2bf(siluf(acc0[r] + bias));
      hidS[(16+M)*128   + nt*16 + nl] = f2bf(siluf(acc1[r] + bias));
    }
  }
  __syncthreads();

  // ---- phase 4: edge_feat = silu(hidden1 @ We2 + be2); agg[row] += edge_feat ----
  for (int nt = 0; nt < 8; ++nt) {
    v8f acc0 = {}, acc1 = {};
    #pragma unroll
    for (int kt = 0; kt < 4; ++kt) {
      v16bf b = load_b_frag(frags, OFF_WE2 + kt*8 + nt, lane);
      acc0 = wmma_bf16(load_a_lds(hidS,          128, kt, lane), b, acc0);
      acc1 = wmma_bf16(load_a_lds(hidS + 16*128, 128, kt, lane), b, acc1);
    }
    const float bias = be2[nt*16 + nl];
    #pragma unroll
    for (int r = 0; r < 8; ++r) {
      const int M = r + 8*half;
      const float v0 = siluf(acc0[r] + bias);
      const float v1 = siluf(acc1[r] + bias);
      einS[M*128      + nt*16 + nl] = f2bf(v0);    // reuse ein buffer as edge_feat
      einS[(16+M)*128 + nt*16 + nl] = f2bf(v1);
      if (ebase + M < E_)
        atomicAdd(&agg[(size_t)eRow[wave][M]*128 + nt*16 + nl], v0);
      if (ebase + 16 + M < E_)
        atomicAdd(&agg[(size_t)eRow[wave][16+M]*128 + nt*16 + nl], v1);
    }
  }
  __syncthreads();

  // ---- phase 5: coord head: dot(silu(feat @ Wc1 + bc1), wc2) -> trans atomics ----
  {
    float pc0[8], pc1[8];
    #pragma unroll
    for (int r = 0; r < 8; ++r) { pc0[r] = 0.f; pc1[r] = 0.f; }
    for (int nt = 0; nt < 8; ++nt) {
      v8f acc0 = {}, acc1 = {};
      #pragma unroll
      for (int kt = 0; kt < 4; ++kt) {
        v16bf b = load_b_frag(frags, OFF_WC1 + kt*8 + nt, lane);
        acc0 = wmma_bf16(load_a_lds(einS,          128, kt, lane), b, acc0);
        acc1 = wmma_bf16(load_a_lds(einS + 16*128, 128, kt, lane), b, acc1);
      }
      const float bias = bc1[nt*16 + nl];
      const float w = shc2[nt*16 + nl];
      #pragma unroll
      for (int r = 0; r < 8; ++r) {
        pc0[r] += siluf(acc0[r] + bias) * w;
        pc1[r] += siluf(acc1[r] + bias) * w;
      }
    }
    #pragma unroll
    for (int s = 1; s < 16; s <<= 1)
      #pragma unroll
      for (int r = 0; r < 8; ++r) {
        pc0[r] += __shfl_xor(pc0[r], s, 32);
        pc1[r] += __shfl_xor(pc1[r], s, 32);
      }
    if (nl < 8) {
      const int m = nl + 8*half;
      float s0 = 0.f, s1 = 0.f;
      #pragma unroll
      for (int r = 0; r < 8; ++r) {
        s0 = (nl == r) ? pc0[r] : s0;
        s1 = (nl == r) ? pc1[r] : s1;
      }
      #pragma unroll
      for (int mt = 0; mt < 2; ++mt) {
        const int ml = mt*16 + m;
        const float sd = mt ? s1 : s0;
        if (ebase + ml < E_) {
          const int rr = eRow[wave][ml];
          atomicAdd(&transAcc[(size_t)rr*4+0], cdS[wave][ml][0]*sd);
          atomicAdd(&transAcc[(size_t)rr*4+1], cdS[wave][ml][1]*sd);
          atomicAdd(&transAcc[(size_t)rr*4+2], cdS[wave][ml][2]*sd);
          atomicAdd(&transAcc[(size_t)rr*4+3], 1.0f);      // segment count
        }
      }
    }
  }

  // ---- phase 6: quat head: silu(feat @ Wq1 + bq1) @ Wq2 -> tan atomics ----
  {
    float p0[2][8], p1[2][8], p2[2][8];
    #pragma unroll
    for (int mt = 0; mt < 2; ++mt)
      #pragma unroll
      for (int r = 0; r < 8; ++r) { p0[mt][r] = 0.f; p1[mt][r] = 0.f; p2[mt][r] = 0.f; }
    for (int nt = 0; nt < 8; ++nt) {
      v8f acc0 = {}, acc1 = {};
      #pragma unroll
      for (int kt = 0; kt < 4; ++kt) {
        v16bf b = load_b_frag(frags, OFF_WQ1 + kt*8 + nt, lane);
        acc0 = wmma_bf16(load_a_lds(einS,          128, kt, lane), b, acc0);
        acc1 = wmma_bf16(load_a_lds(einS + 16*128, 128, kt, lane), b, acc1);
      }
      const float bias = bq1[nt*16 + nl];
      const int N = nt*16 + nl;
      const float w0 = shq2[N*3+0], w1 = shq2[N*3+1], w2 = shq2[N*3+2];
      #pragma unroll
      for (int r = 0; r < 8; ++r) {
        const float v0 = siluf(acc0[r] + bias);
        const float v1 = siluf(acc1[r] + bias);
        p0[0][r] += v0*w0; p1[0][r] += v0*w1; p2[0][r] += v0*w2;
        p0[1][r] += v1*w0; p1[1][r] += v1*w1; p2[1][r] += v1*w2;
      }
    }
    #pragma unroll
    for (int s = 1; s < 16; s <<= 1)
      #pragma unroll
      for (int mt = 0; mt < 2; ++mt)
        #pragma unroll
        for (int r = 0; r < 8; ++r) {
          p0[mt][r] += __shfl_xor(p0[mt][r], s, 32);
          p1[mt][r] += __shfl_xor(p1[mt][r], s, 32);
          p2[mt][r] += __shfl_xor(p2[mt][r], s, 32);
        }
    if (nl < 8) {
      const int m = nl + 8*half;
      #pragma unroll
      for (int mt = 0; mt < 2; ++mt) {
        float s0 = 0.f, s1 = 0.f, s2 = 0.f;
        #pragma unroll
        for (int r = 0; r < 8; ++r) {
          s0 = (nl == r) ? p0[mt][r] : s0;
          s1 = (nl == r) ? p1[mt][r] : s1;
          s2 = (nl == r) ? p2[mt][r] : s2;
        }
        const int ml = mt*16 + m;
        if (ebase + ml < E_) {
          const int rr = eRow[wave][ml];
          atomicAdd(&tanAcc[(size_t)rr*4+0], s0);
          atomicAdd(&tanAcc[(size_t)rr*4+1], s1);
          atomicAdd(&tanAcc[(size_t)rr*4+2], s2);
        }
      }
    }
  }
}

// ---------------------------------------------------------------------------
// Node kernel: coord/quat finalize + residual node MLP on WMMA.
__global__ __launch_bounds__(128) void fegcl_node(
    const float* __restrict__ h, const float* __restrict__ coord,
    const float* __restrict__ quat,
    const float* __restrict__ bn1, const float* __restrict__ bn2,
    const float* __restrict__ bq2,
    const unsigned short* __restrict__ frags,
    const float* __restrict__ agg, const float* __restrict__ transAcc,
    const float* __restrict__ tanAcc,
    float* __restrict__ out, int Nn) {
  __shared__ unsigned int sN_[NWAVES][1536];  // 16x192 bf16 (nin)
  __shared__ unsigned int sH[NWAVES][1024];   // 16x128 bf16 (hidden)
  const int tid = threadIdx.x, wave = tid >> 5, lane = tid & 31;
  const int nl = lane & 15, half = lane >> 4;
  const int nbase = (blockIdx.x * NWAVES + wave) * 16;
  unsigned short* ninS = (unsigned short*)sN_[wave];
  unsigned short* hidS = (unsigned short*)sH[wave];
  float* hout = out;
  float* cout = out + (size_t)Nn * 64;
  float* qout = out + (size_t)Nn * 67;

  // ---- coord / quat (lanes 0..15, one node each) ----
  if (lane < 16) {
    const int n0 = nbase + lane;
    const int n = min(n0, Nn - 1);
    const float cnt = transAcc[(size_t)n*4+3];
    const float invc = rcpf(fmaxf(cnt, 1.0f));
    const float cx = coord[n*3+0] + transAcc[(size_t)n*4+0]*invc;
    const float cy = coord[n*3+1] + transAcc[(size_t)n*4+1]*invc;
    const float cz = coord[n*3+2] + transAcc[(size_t)n*4+2]*invc;
    // mean(tan + bq2) == (sum + cnt*bq2)/max(cnt,1)
    const float txm = (tanAcc[(size_t)n*4+0] + bq2[0]*cnt)*invc;
    const float tym = (tanAcc[(size_t)n*4+1] + bq2[1]*cnt)*invc;
    const float tzm = (tanAcc[(size_t)n*4+2] + bq2[2]*cnt)*invc;
    const float rr = sqrtf(txm*txm + tym*tym + tzm*tzm);
    const float sc = sinf(rr) * rcpf(rr);
    const float rx = txm*sc, ry = tym*sc, rz = tzm*sc, rw = cosf(rr);
    const float px = quat[n*4], py = quat[n*4+1], pz = quat[n*4+2], pw = quat[n*4+3];
    const float ox = pw*rx + px*rw + py*rz - pz*ry;
    const float oy = pw*ry - px*rz + py*rw + pz*rx;
    const float oz = pw*rz + px*ry - py*rx + pz*rw;
    const float ow = pw*rw - px*rx - py*ry - pz*rz;
    if (n0 < Nn) {
      cout[n*3+0] = cx; cout[n*3+1] = cy; cout[n*3+2] = cz;
      qout[n*4+0] = ox; qout[n*4+1] = oy; qout[n*4+2] = oz; qout[n*4+3] = ow;
    }
  }

  // ---- build nin = [h | agg] in bf16 (fully coalesced: consecutive nodes) ----
  for (int i = lane; i < 256; i += 32) {
    int m = i >> 4, f4 = (i & 15) * 4;
    int n = min(nbase + m, Nn - 1);
    float4 v = *(const float4*)(h + (size_t)n*64 + f4);
    unsigned int* np = (unsigned int*)(ninS + m*192);
    np[(f4>>1)+0] = pack2bf(v.x, v.y);
    np[(f4>>1)+1] = pack2bf(v.z, v.w);
  }
  for (int i = lane; i < 512; i += 32) {
    int m = i >> 5, f4 = (i & 31) * 4;
    int n = min(nbase + m, Nn - 1);
    float4 v = *(const float4*)(agg + (size_t)n*128 + f4);
    unsigned int* np = (unsigned int*)(ninS + m*192 + 64);
    np[(f4>>1)+0] = pack2bf(v.x, v.y);
    np[(f4>>1)+1] = pack2bf(v.z, v.w);
  }
  __syncthreads();

  // ---- hidden = silu(nin @ Wn1 + bn1) ----
  for (int nt = 0; nt < 8; ++nt) {
    v8f acc = {};
    #pragma unroll
    for (int kt = 0; kt < 6; ++kt)
      acc = wmma_bf16(load_a_lds(ninS, 192, kt, lane),
                      load_b_frag(frags, OFF_WN1 + kt*8 + nt, lane), acc);
    const float bias = bn1[nt*16 + nl];
    #pragma unroll
    for (int r = 0; r < 8; ++r)
      hidS[(r + 8*half)*128 + nt*16 + nl] = f2bf(siluf(acc[r] + bias));
  }
  __syncthreads();

  // ---- h_out = h + hidden @ Wn2 + bn2 ----
  for (int nt = 0; nt < 4; ++nt) {
    v8f acc = {};
    #pragma unroll
    for (int kt = 0; kt < 4; ++kt)
      acc = wmma_bf16(load_a_lds(hidS, 128, kt, lane),
                      load_b_frag(frags, OFF_WN2 + kt*4 + nt, lane), acc);
    const float bias = bn2[nt*16 + nl];
    #pragma unroll
    for (int r = 0; r < 8; ++r) {
      const int M = r + 8*half;
      const int n0 = nbase + M;
      if (n0 < Nn) {
        const int N = nt*16 + nl;
        hout[(size_t)n0*64 + N] = h[(size_t)n0*64 + N] + acc[r] + bias;
      }
    }
  }
}

// ---------------------------------------------------------------------------
extern "C" void kernel_launch(void* const* d_in, const int* in_sizes, int n_in,
                              void* d_out, int out_size, void* d_ws, size_t ws_size,
                              hipStream_t stream) {
  const float* h     = (const float*)d_in[0];
  const int*   ei    = (const int*)  d_in[1];
  const float* coord = (const float*)d_in[2];
  const float* quat  = (const float*)d_in[3];
  const float* We1 = (const float*)d_in[4];  const float* be1 = (const float*)d_in[5];
  const float* We2 = (const float*)d_in[6];  const float* be2 = (const float*)d_in[7];
  const float* Wn1 = (const float*)d_in[8];  const float* bn1 = (const float*)d_in[9];
  const float* Wn2 = (const float*)d_in[10]; const float* bn2 = (const float*)d_in[11];
  const float* Wc1 = (const float*)d_in[12]; const float* bc1 = (const float*)d_in[13];
  const float* wc2 = (const float*)d_in[14];
  const float* Wq1 = (const float*)d_in[15]; const float* bq1 = (const float*)d_in[16];
  const float* Wq2 = (const float*)d_in[17]; const float* bq2 = (const float*)d_in[18];

  const int Nn = in_sizes[0] / 64;
  const int Ee = in_sizes[1] / 2;

  float* ws = (float*)d_ws;
  float* agg      = ws;                              // Nn*128 f32
  float* transAcc = agg + (size_t)Nn * 128;          // Nn*4  f32 (x,y,z,count)
  float* tanAcc   = transAcc + (size_t)Nn * 4;       // Nn*4  f32
  unsigned short* frags = (unsigned short*)(tanAcc + (size_t)Nn * 4);

  // zero the accumulators (harness does not re-initialize d_ws between calls)
  const size_t zn = (size_t)Nn * 136;
  fegcl_zero<<<2048, 256, 0, stream>>>(ws, zn);

  // bf16 fragment prep for all weight matrices
  fegcl_prep<<<NFRAGS, 512, 0, stream>>>(We1, We2, Wc1, Wq1, Wn1, Wn2, frags);

  // edge pass: 32 edges per wave, 2 waves per block
  const int etiles = (Ee + 31) / 32;
  const int eblocks = (etiles + EWAVES - 1) / EWAVES;
  fegcl_edge<<<eblocks, 64, 0, stream>>>(h, ei, coord, quat, be1, be2, bc1, wc2,
                                         bq1, Wq2, frags, agg, transAcc, tanAcc, Ee);

  // node pass
  const int ntiles = (Nn + 15) / 16;
  const int nblocks = (ntiles + NWAVES - 1) / NWAVES;
  fegcl_node<<<nblocks, 128, 0, stream>>>(h, coord, quat, bn1, bn2, bq2, frags,
                                          agg, transAcc, tanAcc, (float*)d_out, Nn);
}
